// Attention_65807488909977
// MI455X (gfx1250) — matrix operand order repaired and checked by reading, not compiled, not run on hardware
//
#include <hip/hip_runtime.h>
#include <math.h>

// ---------------------------------------------------------------------------
// Bilinear attention, fused pipeline for MI455X (gfx1250, wave32, WMMA).
//   qW = Q @ W ; S = qW @ K^T + bias ; P = softmax(S) ; ctx = P @ V
// Outputs: weights P (fp32, B*LQ*LK) then ctx (fp32, B*LQ*DV), concatenated.
// All GEMMs use v_wmma_f32_16x16x32_bf16 (fp32 accumulate), K-loop software
// pipelined (2 fragment sets), fp32->bf16 on the fly via v_perm_b32 packing.
// ---------------------------------------------------------------------------

#define B_  8
#define LQ_ 2048
#define LK_ 2048
#define D_  1024
#define DV_ 1024

typedef __attribute__((ext_vector_type(16))) __bf16         v16bf;
typedef __attribute__((ext_vector_type(8)))  float          v8f;
typedef __attribute__((ext_vector_type(4)))  float          v4f;
typedef __attribute__((ext_vector_type(16))) unsigned short v16u16;
typedef __attribute__((ext_vector_type(8)))  unsigned short v8u16;
typedef __attribute__((ext_vector_type(2)))  unsigned int   v2u32;

// Two fp32 -> one dword of two bf16 (truncation) with a single v_perm_b32.
__device__ __forceinline__ unsigned int pk_bf16(float lo, float hi) {
    union { float f; unsigned int u; } a, b;
    a.f = hi; b.f = lo;
    // result bytes: {hi[3],hi[2],lo[3],lo[2]} -> sel 0x07060302
    return __builtin_amdgcn_perm(a.u, b.u, 0x07060302u);
}

// fp32 -> bf16 bits, round-to-nearest-even (used in memory-bound prep kernels)
__device__ __forceinline__ unsigned short f2bfu(float f) {
    union { float f; unsigned int u; } c; c.f = f;
    unsigned int u = c.u;
    return (unsigned short)((u + 0x7FFFu + ((u >> 16) & 1u)) >> 16);
}

union Frag16 { v16u16 v; v8u16 h[2]; unsigned int d[8]; unsigned short e[16]; };

// ---- WMMA wrapper: D = A(16x32 bf16) * B(32x16 bf16) + C(16x16 f32) -------
__device__ __forceinline__ v8f wmma_bf16(v16u16 a, v16u16 b, v8f c) {
    union { v16u16 u; v16bf b; } ua, ub;
    ua.u = a; ub.u = b;
    return __builtin_amdgcn_wmma_f32_16x16x32_bf16(
        /*neg_a=*/false, ua.b, /*neg_b=*/false, ub.b,
        /*c_mod=*/(short)0, c, /*reuse_a=*/false, /*reuse_b=*/false);
}

// ---- A fragment (16x32, row-major MxK). Per ISA layout:
//   lane = m (0..15) in each half; half = lane>>4
//   e[0..7]  = A[m][k0 + half*8 + 0..7]
//   e[8..15] = A[m][k0 + 16 + half*8 + 0..7]
__device__ __forceinline__ v16u16 load_a_frag(const unsigned short* __restrict__ A,
                                              int lda, int row, int k0, int lane) {
    const int m = lane & 15, h = lane >> 4;
    const unsigned short* p = A + (size_t)(row + m) * lda + k0 + h * 8;
    Frag16 f;
    f.h[0] = *(const v8u16*)(p);
    f.h[1] = *(const v8u16*)(p + 16);
    return f.v;
}

__device__ __forceinline__ v16u16 load_a_frag(const float* __restrict__ A,
                                              int lda, int row, int k0, int lane) {
    const int m = lane & 15, h = lane >> 4;
    const float* p = A + (size_t)(row + m) * lda + k0 + h * 8;
    v4f f0 = *(const v4f*)(p);      v4f f1 = *(const v4f*)(p + 4);
    v4f f2 = *(const v4f*)(p + 16); v4f f3 = *(const v4f*)(p + 20);
    Frag16 f;
    f.d[0] = pk_bf16(f0[0], f0[1]); f.d[1] = pk_bf16(f0[2], f0[3]);
    f.d[2] = pk_bf16(f1[0], f1[1]); f.d[3] = pk_bf16(f1[2], f1[3]);
    f.d[4] = pk_bf16(f2[0], f2[1]); f.d[5] = pk_bf16(f2[2], f2[3]);
    f.d[6] = pk_bf16(f3[0], f3[1]); f.d[7] = pk_bf16(f3[2], f3[3]);
    return f.v;
}

// ---- B fragment (32x16 KxN) from Bt stored [N][K]:
//   lane = n (0..15) in each half; half = lane>>4
//   e[0..7]  = Bt[n][k0 + half*8 + 0..7], e[8..15] = Bt[n][k0 + 16 + half*8 + 0..7]
__device__ __forceinline__ v16u16 load_b_frag(const unsigned short* __restrict__ Bt,
                                              int ldb, int col, int k0, int lane) {
    const int n = lane & 15, h = lane >> 4;
    const unsigned short* p = Bt + (size_t)(col + n) * ldb + k0 + h * 8;
    Frag16 f;
    f.h[0] = *(const v8u16*)(p);
    f.h[1] = *(const v8u16*)(p + 16);
    return f.v;
}

__device__ __forceinline__ void store_c(float v, float* p)          { *p = v; }
__device__ __forceinline__ void store_c(float v, unsigned short* p) { *p = f2bfu(v); }

// ---------------------------------------------------------------------------
// Generic batched GEMM:  C[b] (MxN, row-major) = A[b] (MxK) * Bt[b] (NxK)^T
// Block: 256 thr = 8 waves; block tile 128x128; wave tile 64x32 (4x2 WMMA).
// K-loop unrolled x2 with two fragment sets so loads of set1 overlap WMMAs
// of set0 (K must be a multiple of 64 -- true for all call sites).
// ---------------------------------------------------------------------------
template <typename AT, typename OT, bool BIAS>
__global__ __launch_bounds__(256)
void gemm_wmma_bf16(const AT* __restrict__ A, const unsigned short* __restrict__ Bt,
                    OT* __restrict__ C, int M, int N, int K,
                    long long sA, long long sB, long long sC,
                    const float* __restrict__ bias) {
    A  += (size_t)blockIdx.z * sA;
    Bt += (size_t)blockIdx.z * sB;
    C  += (size_t)blockIdx.z * sC;

    const int tid  = threadIdx.x;
    const int lane = tid & 31;
    const int wave = tid >> 5;
    const int wm   = wave >> 2;   // 0..1
    const int wn   = wave & 3;    // 0..3
    const int rowBase = blockIdx.y * 128 + wm * 64;
    const int colBase = blockIdx.x * 128 + wn * 32;
    const float bv = BIAS ? bias[0] : 0.0f;

    v8f acc[4][2];
#pragma unroll
    for (int mt = 0; mt < 4; ++mt)
#pragma unroll
        for (int nt = 0; nt < 2; ++nt) acc[mt][nt] = (v8f)(0.0f);

    v16u16 a0[4], b0[2], a1[4], b1[2];

    // prologue: set0 at k=0
#pragma unroll
    for (int mt = 0; mt < 4; ++mt) a0[mt] = load_a_frag(A, K, rowBase + mt * 16, 0, lane);
#pragma unroll
    for (int nt = 0; nt < 2; ++nt) b0[nt] = load_b_frag(Bt, K, colBase + nt * 16, 0, lane);

    for (int k0 = 0; k0 < K; k0 += 64) {
        // issue set1 loads (k0+32) while set0 WMMAs execute
#pragma unroll
        for (int mt = 0; mt < 4; ++mt)
            a1[mt] = load_a_frag(A, K, rowBase + mt * 16, k0 + 32, lane);
#pragma unroll
        for (int nt = 0; nt < 2; ++nt)
            b1[nt] = load_b_frag(Bt, K, colBase + nt * 16, k0 + 32, lane);
#pragma unroll
        for (int mt = 0; mt < 4; ++mt)
#pragma unroll
            for (int nt = 0; nt < 2; ++nt)
                acc[mt][nt] = wmma_bf16(a0[mt], b0[nt], acc[mt][nt]);

        // issue set0 loads (k0+64) while set1 WMMAs execute
        if (k0 + 64 < K) {
#pragma unroll
            for (int mt = 0; mt < 4; ++mt)
                a0[mt] = load_a_frag(A, K, rowBase + mt * 16, k0 + 64, lane);
#pragma unroll
            for (int nt = 0; nt < 2; ++nt)
                b0[nt] = load_b_frag(Bt, K, colBase + nt * 16, k0 + 64, lane);
        }
#pragma unroll
        for (int mt = 0; mt < 4; ++mt)
#pragma unroll
            for (int nt = 0; nt < 2; ++nt)
                acc[mt][nt] = wmma_bf16(a1[mt], b1[nt], acc[mt][nt]);
    }

    // C/D layout: lane half h=lane>>4, n=lane&15; VGPR r -> row (h*8 + r)
    const int n = lane & 15, h = lane >> 4;
#pragma unroll
    for (int mt = 0; mt < 4; ++mt)
#pragma unroll
        for (int nt = 0; nt < 2; ++nt)
#pragma unroll
            for (int r = 0; r < 8; ++r) {
                const int row = rowBase + mt * 16 + h * 8 + r;
                const int col = colBase + nt * 16 + n;
                store_c(acc[mt][nt][r] + bv, C + (size_t)row * N + col);
            }
}

// ---------------------------------------------------------------------------
// Elementwise fp32 -> bf16 convert, 4 elements/thread/iter (b128 -> b64)
// ---------------------------------------------------------------------------
__global__ __launch_bounds__(256)
void cvt_f32_bf16_v4(const float* __restrict__ in, unsigned int* __restrict__ out,
                     size_t n4) {
    size_t i = (size_t)blockIdx.x * 256 + threadIdx.x;
    const size_t stride = (size_t)gridDim.x * 256;
    for (; i < n4; i += stride) {
        v4f f = ((const v4f*)in)[i];
        v2u32 o;
        o.x = pk_bf16(f[0], f[1]);
        o.y = pk_bf16(f[2], f[3]);
        ((v2u32*)out)[i] = o;
    }
}

// ---------------------------------------------------------------------------
// Tiled transpose + convert: out[c][r] (bf16) = in[r][c] (fp32); 32x32 tiles.
// ---------------------------------------------------------------------------
__global__ __launch_bounds__(256)
void transpose_cvt_bf16(const float* __restrict__ in, unsigned short* __restrict__ out,
                        int R, int Cdim, long long sIn, long long sOut) {
    __shared__ float tile[32][33];
    in  += (size_t)blockIdx.z * sIn;
    out += (size_t)blockIdx.z * sOut;
    const int r0 = blockIdx.y * 32, c0 = blockIdx.x * 32;
    const int tx = threadIdx.x & 31;
    const int ty = threadIdx.x >> 5;     // 0..7
#pragma unroll
    for (int i = 0; i < 4; ++i) {
        const int r = ty + i * 8;
        tile[r][tx] = in[(size_t)(r0 + r) * Cdim + c0 + tx];
    }
    __syncthreads();
#pragma unroll
    for (int i = 0; i < 4; ++i) {
        const int c = ty + i * 8;
        out[(size_t)(c0 + c) * R + r0 + tx] = f2bfu(tile[tx][c]);
    }
}

// ---------------------------------------------------------------------------
// In-place row softmax over width 2048, one 256-thread block per row.
// ---------------------------------------------------------------------------
__global__ __launch_bounds__(256)
void softmax_row2048(float* __restrict__ s) {
    __shared__ float red[256];
    float* p = s + (size_t)blockIdx.x * LK_;
    const int t = threadIdx.x;

    float vals[8];
    float vmax = -INFINITY;
#pragma unroll
    for (int i = 0; i < 8; ++i) {
        vals[i] = p[t + i * 256];
        vmax = fmaxf(vmax, vals[i]);
    }
    red[t] = vmax; __syncthreads();
    for (int off = 128; off > 0; off >>= 1) {
        if (t < off) red[t] = fmaxf(red[t], red[t + off]);
        __syncthreads();
    }
    vmax = red[0]; __syncthreads();

    float sum = 0.0f;
#pragma unroll
    for (int i = 0; i < 8; ++i) {
        vals[i] = __expf(vals[i] - vmax);
        sum += vals[i];
    }
    red[t] = sum; __syncthreads();
    for (int off = 128; off > 0; off >>= 1) {
        if (t < off) red[t] += red[t + off];
        __syncthreads();
    }
    const float inv = 1.0f / red[0];
#pragma unroll
    for (int i = 0; i < 8; ++i) p[t + i * 256] = vals[i] * inv;
}

// ---------------------------------------------------------------------------
extern "C" void kernel_launch(void* const* d_in, const int* in_sizes, int n_in,
                              void* d_out, int out_size, void* d_ws, size_t ws_size,
                              hipStream_t stream) {
    (void)in_sizes; (void)n_in; (void)out_size; (void)ws_size;

    const float* key   = (const float*)d_in[0];   // [B, LK, D]
    const float* query = (const float*)d_in[1];   // [B, LQ, D]
    const float* value = (const float*)d_in[2];   // [B, LK, DV]
    const float* W     = (const float*)d_in[3];   // [D, D]
    const float* bias  = (const float*)d_in[4];   // [1]

    float* weights = (float*)d_out;                        // [B, LQ, LK]
    float* ctx     = weights + (size_t)B_ * LQ_ * LK_;     // [B, LQ, DV]

    // Workspace (bf16 stored as u16):
    unsigned short* kbf = (unsigned short*)d_ws;                 // [B][LK][D]
    unsigned short* Wt  = kbf + (size_t)B_ * LK_ * D_;           // [D][D]   (W^T)
    unsigned short* Vt  = Wt  + (size_t)D_ * D_;                 // [B][DV][LK] (V^T)
    unsigned short* qWb = Vt  + (size_t)B_ * DV_ * LK_;          // [B][LQ][D]

    const dim3 blk(256);

    // 1) key -> bf16 (already [N][K] layout for the scores GEMM)
    cvt_f32_bf16_v4<<<4096, blk, 0, stream>>>(key, (unsigned int*)kbf,
                                              (size_t)B_ * LK_ * D_ / 4);

    // 2) W^T (bf16) and V^T (bf16) for contiguous B-fragment reads
    transpose_cvt_bf16<<<dim3(D_ / 32, D_ / 32, 1), blk, 0, stream>>>(
        W, Wt, D_, D_, 0, 0);
    transpose_cvt_bf16<<<dim3(DV_ / 32, LK_ / 32, B_), blk, 0, stream>>>(
        value, Vt, LK_, DV_, (long long)LK_ * DV_, (long long)DV_ * LK_);

    // 3) qW[b] = Q[b] @ W          (A fp32 on-the-fly, out bf16)  M=LQ N=D K=D
    gemm_wmma_bf16<float, unsigned short, false>
        <<<dim3(D_ / 128, LQ_ / 128, B_), blk, 0, stream>>>(
            query, Wt, qWb, LQ_, D_, D_,
            (long long)LQ_ * D_, 0, (long long)LQ_ * D_, nullptr);

    // 4) S[b] = qW[b] @ K[b]^T + bias   (out fp32 -> weights)    M=LQ N=LK K=D
    gemm_wmma_bf16<unsigned short, float, true>
        <<<dim3(LK_ / 128, LQ_ / 128, B_), blk, 0, stream>>>(
            qWb, kbf, weights, LQ_, LK_, D_,
            (long long)LQ_ * D_, (long long)LK_ * D_, (long long)LQ_ * LK_, bias);

    // 5) softmax rows of weights, in place
    softmax_row2048<<<B_ * LQ_, blk, 0, stream>>>(weights);

    // 6) ctx[b] = P[b] @ V[b]      (A fp32 on-the-fly)           M=LQ N=DV K=LK
    gemm_wmma_bf16<float, float, false>
        <<<dim3(DV_ / 128, LQ_ / 128, B_), blk, 0, stream>>>(
            weights, Vt, ctx, LQ_, LK_, LK_,
            (long long)LQ_ * LK_, (long long)DV_ * LK_, (long long)LQ_ * DV_, nullptr);
}
